// MultiheadAttention_38560216383528
// MI455X (gfx1250) — matrix operand order repaired and checked by reading, not compiled, not run on hardware
//
#include <hip/hip_runtime.h>
#include <hip/hip_bf16.h>
#include <stdint.h>

// ---------------------------------------------------------------------------
// MI455X (gfx1250) MHA forward.
// bf16 operands + f32 WMMA accumulation (v_wmma_f32_16x16x32_bf16).
// Shared operand tiles staged global->LDS with GLOBAL_LOAD_ASYNC_TO_LDS_B128
// (ASYNCcnt), double-buffered so WMMAs overlap the async DMA.
// GEMM waves compute 32x64 tiles (8 WMMA / K-step) to amortize LDS traffic.
// ---------------------------------------------------------------------------

typedef __bf16 bf16_t;
typedef __attribute__((ext_vector_type(16))) __bf16 v16bf;
typedef __attribute__((ext_vector_type(8)))  __bf16 v8bf;
typedef __attribute__((ext_vector_type(8)))  float  v8f;

#define T_LEN   2048
#define BSZ     2
#define EMB     1280
#define NH      20
#define HD      64
#define SCALING 0.125f   // 64^-0.5

#define WMMA_BF16(a,b,c) \
  __builtin_amdgcn_wmma_f32_16x16x32_bf16(false,(a),false,(b),(short)0,(c),false,false)

// ---- async global->LDS staging (ASYNCcnt-tracked) ----
static __device__ __forceinline__ void async_cp16(bf16_t* l, const bf16_t* g) {
  unsigned lds_off = (unsigned)(uintptr_t)l;   // low 32b of shared ptr = LDS offset
  asm volatile("global_load_async_to_lds_b128 %0, %1, off"
               :: "v"(lds_off), "v"(g) : "memory");
}
static __device__ __forceinline__ void async_wait0() {
  asm volatile("s_wait_asynccnt 0" ::: "memory");
}

// A-fragment (16x32 bf16, M x K), row-major src with leading dim ld.
static __device__ __forceinline__ v16bf frag_a(const bf16_t* __restrict__ src,
                                               int ld, int row0, int kb, int lane) {
  int m = lane & 15, hb = lane >> 4;
  const bf16_t* p = src + (size_t)(row0 + m) * ld + kb + hb * 8;
  v8bf lo = *(const v8bf*)p;          // K = hb*8 + 0..7
  v8bf hi = *(const v8bf*)(p + 16);   // K = 16 + hb*8 + 0..7
  v16bf r;
#pragma unroll
  for (int i = 0; i < 8; ++i) { r[i] = lo[i]; r[i + 8] = hi[i]; }
  return r;
}

// B-fragment (32x16 bf16, K x N) from memory laid out [n][k] (K contiguous).
static __device__ __forceinline__ v16bf frag_b(const bf16_t* __restrict__ srcT,
                                               int ld, int n0, int kb, int lane) {
  int n = lane & 15, hb = lane >> 4;
  const bf16_t* p = srcT + (size_t)(n0 + n) * ld + kb + hb * 8;
  v8bf lo = *(const v8bf*)p;
  v8bf hi = *(const v8bf*)(p + 16);
  v16bf r;
#pragma unroll
  for (int i = 0; i < 8; ++i) { r[i] = lo[i]; r[i + 8] = hi[i]; }
  return r;
}

// ---------------------------------------------------------------------------
__global__ void zero_f32(float* __restrict__ p, long long n) {
  long long i = (long long)blockIdx.x * blockDim.x + threadIdx.x;
  long long s = (long long)gridDim.x * blockDim.x;
  for (; i < n; i += s) p[i] = 0.0f;
}

__global__ void cvt_f32_bf16(const float* __restrict__ src, bf16_t* __restrict__ dst,
                             long long n) {
  long long i = (long long)blockIdx.x * blockDim.x + threadIdx.x;
  long long s = (long long)gridDim.x * blockDim.x;
  for (; i < n; i += s) dst[i] = (bf16_t)src[i];
}

// WT[z][n][k] = (bf16) W_z[k][n]
__global__ void cvt_wT(const float* __restrict__ W0, const float* __restrict__ W1,
                       const float* __restrict__ W2, const float* __restrict__ W3,
                       bf16_t* __restrict__ WT) {
  int z = blockIdx.z;
  const float* W = (z == 0) ? W0 : (z == 1) ? W1 : (z == 2) ? W2 : W3;
  bf16_t* D = WT + (size_t)z * EMB * EMB;
  int idx = blockIdx.x * blockDim.x + threadIdx.x;
  if (idx < EMB * EMB) {
    int n = idx / EMB, k = idx % EMB;
    D[idx] = (bf16_t)W[(size_t)k * EMB + n];
  }
}

// ---------------------------------------------------------------------------
// QKV projection. Wave tile = 32x64 (8 WMMA / K-step); block = 256 rows.
//   z=0: Q -> [B*H][T][64] (scaled); z=1: K -> [B*H][T][64]; z=2: Vt -> [B*H][64][T]
__global__ __launch_bounds__(256) void qkv_gemm(
    const bf16_t* __restrict__ X, const bf16_t* __restrict__ WTall,
    const float* __restrict__ bq, const float* __restrict__ bk,
    const float* __restrict__ bv,
    bf16_t* __restrict__ Qbf, bf16_t* __restrict__ Kbf, bf16_t* __restrict__ Vt) {
  int z = blockIdx.z;
  const bf16_t* WT = WTall + (size_t)z * EMB * EMB;
  int tid = threadIdx.x;
  int wave = tid >> 5, lane = tid & 31;
  int row0 = blockIdx.x * 256 + wave * 32;
  int n0b  = blockIdx.y * 64;

  __shared__ __align__(16) bf16_t bt[2][64 * 32];   // [n][k] tile, 4KB each
  int sn = tid >> 2, sk8 = (tid & 3) * 8;           // each thread stages 16B

  async_cp16(&bt[0][sn * 32 + sk8], WT + (size_t)(n0b + sn) * EMB + sk8);

  v8f acc[2][4] = {};
  for (int it = 0; it < EMB / 32; ++it) {
    async_wait0();
    __syncthreads();
    int cur = it & 1;
    if (it + 1 < EMB / 32)
      async_cp16(&bt[cur ^ 1][sn * 32 + sk8],
                 WT + (size_t)(n0b + sn) * EMB + (it + 1) * 32 + sk8);
    int kb = it * 32;
    v16bf a0 = frag_a(X, EMB, row0,      kb, lane);
    v16bf a1 = frag_a(X, EMB, row0 + 16, kb, lane);
    v16bf b0 = frag_b(bt[cur], 32, 0,  0, lane);
    v16bf b1 = frag_b(bt[cur], 32, 16, 0, lane);
    v16bf b2 = frag_b(bt[cur], 32, 32, 0, lane);
    v16bf b3 = frag_b(bt[cur], 32, 48, 0, lane);
    acc[0][0] = WMMA_BF16(a0, b0, acc[0][0]);
    acc[0][1] = WMMA_BF16(a0, b1, acc[0][1]);
    acc[0][2] = WMMA_BF16(a0, b2, acc[0][2]);
    acc[0][3] = WMMA_BF16(a0, b3, acc[0][3]);
    acc[1][0] = WMMA_BF16(a1, b0, acc[1][0]);
    acc[1][1] = WMMA_BF16(a1, b1, acc[1][1]);
    acc[1][2] = WMMA_BF16(a1, b2, acc[1][2]);
    acc[1][3] = WMMA_BF16(a1, b3, acc[1][3]);
  }

  const float* bias = (z == 0) ? bq : (z == 1) ? bk : bv;
  int hb = lane >> 4, nl = lane & 15;
#pragma unroll
  for (int mi = 0; mi < 2; ++mi) {
#pragma unroll
    for (int nt = 0; nt < 4; ++nt) {
#pragma unroll
      for (int r = 0; r < 8; ++r) {
        int row = row0 + mi * 16 + r + 8 * hb;  // = t*BSZ + b
        int col = n0b + nt * 16 + nl;           // = h*64 + d
        float v = acc[mi][nt][r] + bias[col];
        int t = row >> 1, bb = row & 1;
        int h = col >> 6, d = col & 63;
        int bh = bb * NH + h;
        if (z == 0)      Qbf[((size_t)bh * T_LEN + t) * HD + d] = (bf16_t)(v * SCALING);
        else if (z == 1) Kbf[((size_t)bh * T_LEN + t) * HD + d] = (bf16_t)v;
        else             Vt[((size_t)bh * HD + d) * T_LEN + t]  = (bf16_t)v;
      }
    }
  }
}

// ---------------------------------------------------------------------------
// Flash-style attention; K and V chunks async-staged into LDS (shared by all
// 8 waves), double-buffered. Pass 2 re-stages K and scatters avg via atomics.
__global__ __launch_bounds__(256) void attn_kernel(
    const bf16_t* __restrict__ Qbf, const bf16_t* __restrict__ Kbf,
    const bf16_t* __restrict__ Vt, const unsigned char* __restrict__ kpm,
    bf16_t* __restrict__ Obf, float* __restrict__ avg) {
  int bh = blockIdx.x;
  int b = bh / NH, h = bh % NH;
  int tid = threadIdx.x;
  int wave = tid >> 5, lane = tid & 31;
  int hb = lane >> 4, nl = lane & 15;
  int q0 = blockIdx.y * 128 + wave * 16;

  const bf16_t* Qh = Qbf + (size_t)bh * T_LEN * HD;
  const bf16_t* Kh = Kbf + (size_t)bh * T_LEN * HD;
  const bf16_t* Vh = Vt  + (size_t)bh * HD * T_LEN;
  const unsigned char* mk = kpm + (size_t)b * T_LEN;

  __shared__ __align__(16) bf16_t kt[2][32 * 64];     // [key][d], 4KB each
  __shared__ __align__(16) bf16_t vt[2][64 * 32];     // [d][key], 4KB each
  __shared__ __align__(16) bf16_t plds[8][16 * 32];   // per-wave P tile
  bf16_t* my = plds[wave];

  int kkey = tid >> 3, kd8 = (tid & 7) * 8;           // K-tile staging coords
  int vd   = tid >> 2, vk8 = (tid & 3) * 8;           // V-tile staging coords

  v16bf qa0 = frag_a(Qh, HD, q0, 0, lane);
  v16bf qa1 = frag_a(Qh, HD, q0, 32, lane);

  float mrow[8], lrow[8];
#pragma unroll
  for (int r = 0; r < 8; ++r) { mrow[r] = -1e30f; lrow[r] = 0.0f; }
  v8f o[4] = {};

  // ---- pass 1: online softmax + O accumulation ----
  async_cp16(&kt[0][kkey * 64 + kd8], Kh + (size_t)kkey * HD + kd8);
  async_cp16(&vt[0][vd * 32 + vk8],   Vh + (size_t)vd * T_LEN + vk8);

  for (int it = 0; it < T_LEN / 32; ++it) {
    int kc = it * 32;
    async_wait0();
    __syncthreads();
    int cur = it & 1;
    if (it + 1 < T_LEN / 32) {
      int kcn = kc + 32;
      async_cp16(&kt[cur ^ 1][kkey * 64 + kd8], Kh + (size_t)(kcn + kkey) * HD + kd8);
      async_cp16(&vt[cur ^ 1][vd * 32 + vk8],   Vh + (size_t)vd * T_LEN + kcn + vk8);
    }

    v16bf kb00 = frag_b(kt[cur], HD, 0,  0,  lane);
    v16bf kb01 = frag_b(kt[cur], HD, 0,  32, lane);
    v16bf kb10 = frag_b(kt[cur], HD, 16, 0,  lane);
    v16bf kb11 = frag_b(kt[cur], HD, 16, 32, lane);
    v8f c0 = {}, c1 = {};
    c0 = WMMA_BF16(qa0, kb00, c0);
    c0 = WMMA_BF16(qa1, kb01, c0);
    c1 = WMMA_BF16(qa0, kb10, c1);
    c1 = WMMA_BF16(qa1, kb11, c1);

    bool msk0 = mk[kc + nl] != 0;
    bool msk1 = mk[kc + 16 + nl] != 0;

    float scale[8];
#pragma unroll
    for (int r = 0; r < 8; ++r) {
      float s0 = msk0 ? -1e30f : c0[r];
      float s1 = msk1 ? -1e30f : c1[r];
      float mx = fmaxf(s0, s1);
#pragma unroll
      for (int d = 1; d < 16; d <<= 1) mx = fmaxf(mx, __shfl_xor(mx, d, 32));
      float mnew = fmaxf(mrow[r], mx);
      float p0 = msk0 ? 0.0f : __expf(s0 - mnew);
      float p1 = msk1 ? 0.0f : __expf(s1 - mnew);
      float sum = p0 + p1;
#pragma unroll
      for (int d = 1; d < 16; d <<= 1) sum += __shfl_xor(sum, d, 32);
      float sc = __expf(mrow[r] - mnew);
      lrow[r] = lrow[r] * sc + sum;
      mrow[r] = mnew;
      scale[r] = sc;
      int M = r + 8 * hb;
      my[M * 32 + nl]      = (bf16_t)p0;
      my[M * 32 + 16 + nl] = (bf16_t)p1;
    }
    __builtin_amdgcn_wave_barrier();

#pragma unroll
    for (int nt = 0; nt < 4; ++nt)
#pragma unroll
      for (int r = 0; r < 8; ++r) o[nt][r] *= scale[r];

    v16bf pa  = frag_a(my, 32, 0, 0, lane);
    v16bf vb0 = frag_b(vt[cur], 32, 0,  0, lane);
    v16bf vb1 = frag_b(vt[cur], 32, 16, 0, lane);
    v16bf vb2 = frag_b(vt[cur], 32, 32, 0, lane);
    v16bf vb3 = frag_b(vt[cur], 32, 48, 0, lane);
    o[0] = WMMA_BF16(pa, vb0, o[0]);
    o[1] = WMMA_BF16(pa, vb1, o[1]);
    o[2] = WMMA_BF16(pa, vb2, o[2]);
    o[3] = WMMA_BF16(pa, vb3, o[3]);
  }

  // O epilogue -> attn in [T][B][E] bf16 layout
  float linv[8];
#pragma unroll
  for (int r = 0; r < 8; ++r) linv[r] = 1.0f / lrow[r];
#pragma unroll
  for (int nt = 0; nt < 4; ++nt) {
#pragma unroll
    for (int r = 0; r < 8; ++r) {
      int M = r + 8 * hb;
      int t = q0 + M;
      int col = h * HD + nt * 16 + nl;
      Obf[((size_t)t * BSZ + b) * EMB + col] = (bf16_t)(o[nt][r] * linv[r]);
    }
  }

  // ---- pass 2: avg_weights accumulation (re-stage K only) ----
  float* ab = avg + (size_t)b * T_LEN * T_LEN;
  const float invH = 1.0f / (float)NH;
  float pl2[8];
#pragma unroll
  for (int r = 0; r < 8; ++r) pl2[r] = invH * linv[r];

  async_cp16(&kt[0][kkey * 64 + kd8], Kh + (size_t)kkey * HD + kd8);
  for (int it = 0; it < T_LEN / 32; ++it) {
    int kc = it * 32;
    async_wait0();
    __syncthreads();
    int cur = it & 1;
    if (it + 1 < T_LEN / 32)
      async_cp16(&kt[cur ^ 1][kkey * 64 + kd8],
                 Kh + (size_t)(kc + 32 + kkey) * HD + kd8);

    v16bf kb00 = frag_b(kt[cur], HD, 0,  0,  lane);
    v16bf kb01 = frag_b(kt[cur], HD, 0,  32, lane);
    v16bf kb10 = frag_b(kt[cur], HD, 16, 0,  lane);
    v16bf kb11 = frag_b(kt[cur], HD, 16, 32, lane);
    v8f c0 = {}, c1 = {};
    c0 = WMMA_BF16(qa0, kb00, c0);
    c0 = WMMA_BF16(qa1, kb01, c0);
    c1 = WMMA_BF16(qa0, kb10, c1);
    c1 = WMMA_BF16(qa1, kb11, c1);
    bool msk0 = mk[kc + nl] != 0;
    bool msk1 = mk[kc + 16 + nl] != 0;
#pragma unroll
    for (int r = 0; r < 8; ++r) {
      int qi = q0 + r + 8 * hb;
      float p0 = msk0 ? 0.0f : __expf(c0[r] - mrow[r]) * pl2[r];
      float p1 = msk1 ? 0.0f : __expf(c1[r] - mrow[r]) * pl2[r];
      atomicAdd(&ab[(size_t)qi * T_LEN + kc + nl], p0);
      atomicAdd(&ab[(size_t)qi * T_LEN + kc + 16 + nl], p1);
    }
  }
}

// ---------------------------------------------------------------------------
// Output projection. Wave tile = 32x64: attn @ Wo + bo -> fp32 d_out [T,B,E].
__global__ __launch_bounds__(256) void out_gemm(
    const bf16_t* __restrict__ Obf, const bf16_t* __restrict__ WoT,
    const float* __restrict__ bo, float* __restrict__ out) {
  int tid = threadIdx.x;
  int wave = tid >> 5, lane = tid & 31;
  int row0 = blockIdx.x * 256 + wave * 32;
  int n0b  = blockIdx.y * 64;

  __shared__ __align__(16) bf16_t bt[2][64 * 32];
  int sn = tid >> 2, sk8 = (tid & 3) * 8;

  async_cp16(&bt[0][sn * 32 + sk8], WoT + (size_t)(n0b + sn) * EMB + sk8);

  v8f acc[2][4] = {};
  for (int it = 0; it < EMB / 32; ++it) {
    async_wait0();
    __syncthreads();
    int cur = it & 1;
    if (it + 1 < EMB / 32)
      async_cp16(&bt[cur ^ 1][sn * 32 + sk8],
                 WoT + (size_t)(n0b + sn) * EMB + (it + 1) * 32 + sk8);
    int kb = it * 32;
    v16bf a0 = frag_a(Obf, EMB, row0,      kb, lane);
    v16bf a1 = frag_a(Obf, EMB, row0 + 16, kb, lane);
    v16bf b0 = frag_b(bt[cur], 32, 0,  0, lane);
    v16bf b1 = frag_b(bt[cur], 32, 16, 0, lane);
    v16bf b2 = frag_b(bt[cur], 32, 32, 0, lane);
    v16bf b3 = frag_b(bt[cur], 32, 48, 0, lane);
    acc[0][0] = WMMA_BF16(a0, b0, acc[0][0]);
    acc[0][1] = WMMA_BF16(a0, b1, acc[0][1]);
    acc[0][2] = WMMA_BF16(a0, b2, acc[0][2]);
    acc[0][3] = WMMA_BF16(a0, b3, acc[0][3]);
    acc[1][0] = WMMA_BF16(a1, b0, acc[1][0]);
    acc[1][1] = WMMA_BF16(a1, b1, acc[1][1]);
    acc[1][2] = WMMA_BF16(a1, b2, acc[1][2]);
    acc[1][3] = WMMA_BF16(a1, b3, acc[1][3]);
  }

  int hb = lane >> 4, nl = lane & 15;
#pragma unroll
  for (int mi = 0; mi < 2; ++mi)
#pragma unroll
    for (int nt = 0; nt < 4; ++nt)
#pragma unroll
      for (int r = 0; r < 8; ++r) {
        int row = row0 + mi * 16 + r + 8 * hb;
        int col = n0b + nt * 16 + nl;
        out[(size_t)row * EMB + col] = acc[mi][nt][r] + bo[col];
      }
}

// ---------------------------------------------------------------------------
extern "C" void kernel_launch(void* const* d_in, const int* in_sizes, int n_in,
                              void* d_out, int out_size, void* d_ws, size_t ws_size,
                              hipStream_t stream) {
  const float* query = (const float*)d_in[0];
  const unsigned char* kpm = (const unsigned char*)d_in[1];   // bool mask [B,T]
  const float* Wq = (const float*)d_in[2];
  const float* bq = (const float*)d_in[3];
  const float* Wk = (const float*)d_in[4];
  const float* bk = (const float*)d_in[5];
  const float* Wv = (const float*)d_in[6];
  const float* bv = (const float*)d_in[7];
  const float* Wo = (const float*)d_in[8];
  const float* bo = (const float*)d_in[9];

  char* ws = (char*)d_ws;
  bf16_t* Xbf   = (bf16_t*)(ws + 0);          // 10,485,760 B
  bf16_t* WTall = (bf16_t*)(ws + 10485760);   // 13,107,200 B
  bf16_t* Qbf   = (bf16_t*)(ws + 23592960);   // 10,485,760 B
  bf16_t* Kbf   = (bf16_t*)(ws + 34078720);   // 10,485,760 B
  bf16_t* Vtb   = (bf16_t*)(ws + 44564480);   // 10,485,760 B
  bf16_t* Obf   = (bf16_t*)(ws + 55050240);   // 10,485,760 B

  float* out_attn = (float*)d_out;                         // [T,B,E]
  float* out_avg  = out_attn + (size_t)T_LEN * BSZ * EMB;  // [B,T,T]

  zero_f32<<<2048, 256, 0, stream>>>(out_avg, (long long)BSZ * T_LEN * T_LEN);
  cvt_f32_bf16<<<5120, 256, 0, stream>>>(query, Xbf, (long long)T_LEN * BSZ * EMB);
  cvt_wT<<<dim3(6400, 1, 4), 256, 0, stream>>>(Wq, Wk, Wv, Wo, WTall);
  qkv_gemm<<<dim3(16, 20, 3), 256, 0, stream>>>(Xbf, WTall, bq, bk, bv, Qbf, Kbf, Vtb);
  attn_kernel<<<dim3(40, 16), 256, 0, stream>>>(Qbf, Kbf, Vtb, kpm, Obf, out_avg);
  out_gemm<<<dim3(16, 20), 256, 0, stream>>>(Obf, WTall + (size_t)3 * EMB * EMB, bo, out_attn);
}
